// Attention_7138235646274
// MI455X (gfx1250) — matrix operand order repaired and checked by reading, not compiled, not run on hardware
//
#include <hip/hip_runtime.h>
#include <hip/hip_bf16.h>

// MI455X / gfx1250 fused multi-head self-attention.
// T=1024, B=4, E=1024, H=16, D=64.  All matmuls via v_wmma_f32_16x16x32_f16.
// Software-pipelined: next tile's global loads are in flight during compute.

typedef _Float16 half_t;
typedef __attribute__((ext_vector_type(16))) _Float16 v16h;
typedef __attribute__((ext_vector_type(8)))  float    v8f;

#define T_DIM 1024
#define B_DIM 4
#define E_DIM 1024
#define H_DIM 16
#define D_DIM 64
#define BH_DIM 64
#define ROWS (T_DIM * B_DIM)   // 4096

union H8  { _Float16 h[8];  uint4 u; };
union H16 { v16h v; _Float16 h[16]; uint4 u4[2]; };

static __device__ __forceinline__ v8f wmma_f16(const v16h& a, const v16h& b, const v8f& c) {
    return __builtin_amdgcn_wmma_f32_16x16x32_f16(false, a, false, b, (short)0, c, false, false);
}

static __device__ __forceinline__ uint4 cvt8(float4 f0, float4 f1) {
    H8 p;
    p.h[0] = (half_t)f0.x; p.h[1] = (half_t)f0.y;
    p.h[2] = (half_t)f0.z; p.h[3] = (half_t)f0.w;
    p.h[4] = (half_t)f1.x; p.h[5] = (half_t)f1.y;
    p.h[6] = (half_t)f1.z; p.h[7] = (half_t)f1.w;
    return p.u;
}

// ---------------------------------------------------------------------------
// Kernel 1: one projection  dst = (X @ W^T + bias) * scale
//   X: [4096, 1024] fp32 (row = t*B+b), W: [1024,1024] fp32 (row = out feature)
//   TRANSPOSED==0 -> dst[bh][t][dd]   (Q and K)
//   TRANSPOSED==1 -> dst[bh][dd][t]   (V, pre-transposed for PV WMMA B-frags)
// block = 256 threads (8 waves), 128(M) x 64(N) tile, K chunks of 64,
// next chunk's global loads staged in registers during WMMA compute.
// ---------------------------------------------------------------------------
template<int TRANSPOSED>
__global__ __launch_bounds__(256)
void qkv_proj_kernel(const float* __restrict__ X,
                     const float* __restrict__ W,
                     const float* __restrict__ bias,
                     half_t* __restrict__ dst,
                     float scale)
{
    __shared__ half_t Xs[128 * 64];   // [m][k] fp16, 16 KB
    __shared__ half_t Ws[64 * 64];    // [n][k] fp16,  8 KB

    const int tid  = threadIdx.x;
    const int wave = tid >> 5;        // 0..7
    const int lane = tid & 31;
    const int colL = lane & 15;
    const bool hi  = lane >= 16;

    const int m0 = blockIdx.y * 128;
    const int n0 = blockIdx.x * 64;

    const v8f vzero = {0.f,0.f,0.f,0.f,0.f,0.f,0.f,0.f};
    v8f acc[4];
#pragma unroll
    for (int j = 0; j < 4; ++j) acc[j] = vzero;

    const int ldrX = tid >> 1;            // 0..127
    const int ldcX = (tid & 1) * 32;      // 0 or 32
    const int ldrW = tid >> 2;            // 0..63
    const int ldcW = (tid & 3) * 16;      // 0,16,32,48

    float4 xst[8];   // staged X: 32 floats
    float4 wst[4];   // staged W: 16 floats

    auto stage = [&](int kc) {
        const float* xsrc = X + (size_t)(m0 + ldrX) * E_DIM + kc * 64 + ldcX;
#pragma unroll
        for (int g = 0; g < 8; ++g) xst[g] = *(const float4*)(xsrc + g * 4);
        const float* wsrc = W + (size_t)(n0 + ldrW) * E_DIM + kc * 64 + ldcW;
#pragma unroll
        for (int g = 0; g < 4; ++g) wst[g] = *(const float4*)(wsrc + g * 4);
    };
    auto commit = [&]() {
#pragma unroll
        for (int g = 0; g < 4; ++g)
            *(uint4*)&Xs[ldrX * 64 + ldcX + g * 8] = cvt8(xst[2 * g], xst[2 * g + 1]);
#pragma unroll
        for (int g = 0; g < 2; ++g)
            *(uint4*)&Ws[ldrW * 64 + ldcW + g * 8] = cvt8(wst[2 * g], wst[2 * g + 1]);
    };

    stage(0);
    for (int kc = 0; kc < E_DIM / 64; ++kc) {
        commit();
        __syncthreads();
        if (kc + 1 < E_DIM / 64) stage(kc + 1);   // in flight during compute

        // ---- A fragments: two 16x32 (k-halves of the 64-chunk) ----
        const int mA  = wave * 16 + colL;
        const int kbA = hi ? 8 : 0;
        H16 a[2];
#pragma unroll
        for (int c = 0; c < 2; ++c) {
            a[c].u4[0] = *(const uint4*)&Xs[mA * 64 + c * 32 + kbA];
            a[c].u4[1] = *(const uint4*)&Xs[mA * 64 + c * 32 + kbA + 16];
        }
        // ---- 8 B fragments + 8 WMMAs ----
        H16 bf[4][2];
#pragma unroll
        for (int j = 0; j < 4; ++j) {
            const int nB = j * 16 + colL;
#pragma unroll
            for (int c = 0; c < 2; ++c) {
                const int kbB = c * 32 + (hi ? 8 : 0);
                bf[j][c].u4[0] = *(const uint4*)&Ws[nB * 64 + kbB];
                bf[j][c].u4[1] = *(const uint4*)&Ws[nB * 64 + kbB + 16];
            }
        }
#pragma unroll
        for (int j = 0; j < 4; ++j) {
            acc[j] = wmma_f16(a[0].v, bf[j][0].v, acc[j]);
            acc[j] = wmma_f16(a[1].v, bf[j][1].v, acc[j]);
        }
        __syncthreads();
    }

    // ---- epilogue: bias, scale, scatter as fp16 into head-major layout ----
#pragma unroll
    for (int j = 0; j < 4; ++j) {
        const int n  = n0 + j * 16 + colL;     // output feature
        const int h  = n >> 6;
        const int dd = n & 63;
        const float bv = bias[n];
#pragma unroll
        for (int r = 0; r < 8; ++r) {
            const int i = m0 + wave * 16 + r + (hi ? 8 : 0);   // row in [4096)
            const int t = i >> 2;
            const int b = i & 3;
            const int bh = b * H_DIM + h;
            const float val = (acc[j][r] + bv) * scale;
            if (TRANSPOSED)
                dst[(size_t)(bh * D_DIM + dd) * T_DIM + t] = (half_t)val;
            else
                dst[(size_t)(bh * T_DIM + t) * D_DIM + dd] = (half_t)val;
        }
    }
}

// ---------------------------------------------------------------------------
// Kernel 2: flash-style causal attention per head.
// grid = (BH=64, T/64=16); block = 128 threads (4 waves), each wave owns a
// 16-row Q tile. Online softmax over causal key chunks of 64; next chunk's
// K/V global loads staged in registers during the current chunk's compute.
// ---------------------------------------------------------------------------
__global__ __launch_bounds__(128)
void attn_kernel(const half_t* __restrict__ Qh,   // [bh][t][64]
                 const half_t* __restrict__ Kh,   // [bh][t][64]
                 const half_t* __restrict__ Vt,   // [bh][64][t]
                 const unsigned char* __restrict__ padmask, // [B][T] bool
                 float* __restrict__ out)         // [T][B][E] fp32
{
    __shared__ half_t Ks[64 * 64];        // [s_local][dd]
    __shared__ half_t Vs[64 * 64];        // [dd][s_local]
    __shared__ half_t Ps[4 * 16 * 64];    // per-wave P tile [row][s_local]

    const int bh = blockIdx.x;
    const int tb = blockIdx.y;
    const int b  = bh >> 4;
    const int h  = bh & 15;

    const int tid  = threadIdx.x;
    const int wave = tid >> 5;
    const int lane = tid & 31;
    const int colL = lane & 15;
    const bool hi  = lane >= 16;
    const int kbA  = hi ? 8 : 0;

    const int q0 = tb * 64 + wave * 16;   // wave's first query row

    // ---- Q A-fragments (16x32 x2), straight from global in A layout ----
    const half_t* qbase = Qh + (size_t)(bh * T_DIM + q0 + colL) * D_DIM;
    H16 qf[2];
#pragma unroll
    for (int c = 0; c < 2; ++c) {
        qf[c].u4[0] = *(const uint4*)&qbase[c * 32 + kbA];
        qf[c].u4[1] = *(const uint4*)&qbase[c * 32 + kbA + 16];
    }

    const v8f vzero = {0.f,0.f,0.f,0.f,0.f,0.f,0.f,0.f};
    v8f O[4];
#pragma unroll
    for (int j = 0; j < 4; ++j) O[j] = vzero;
    float m[8], l[8];
#pragma unroll
    for (int r = 0; r < 8; ++r) { m[r] = -1e30f; l[r] = 0.f; }

    const int ldr = tid >> 1;          // 0..63
    const int ldc = (tid & 1) * 32;    // 0 or 32

    uint4 kst[4], vst[4];              // staged K/V rows for next chunk
    auto stage = [&](int s0) {
        const uint4* ksrc = (const uint4*)(Kh + (size_t)(bh * T_DIM + s0 + ldr) * D_DIM + ldc);
#pragma unroll
        for (int g = 0; g < 4; ++g) kst[g] = ksrc[g];
        const uint4* vsrc = (const uint4*)(Vt + (size_t)(bh * D_DIM + ldr) * T_DIM + s0 + ldc);
#pragma unroll
        for (int g = 0; g < 4; ++g) vst[g] = vsrc[g];
    };
    auto commit = [&]() {
        uint4* kdst = (uint4*)&Ks[ldr * 64 + ldc];
#pragma unroll
        for (int g = 0; g < 4; ++g) kdst[g] = kst[g];
        uint4* vdst = (uint4*)&Vs[ldr * 64 + ldc];
#pragma unroll
        for (int g = 0; g < 4; ++g) vdst[g] = vst[g];
    };

    const int nchunks = tb + 1;        // causal: keys only up to this row block
    stage(0);
    for (int ch = 0; ch < nchunks; ++ch) {
        const int s0 = ch * 64;
        commit();
        __syncthreads();
        if (ch + 1 < nchunks) stage(s0 + 64);   // in flight during compute

        // ---- S = Q K^T  (16 x 64 per wave): 8 B-frags, 8 WMMAs ----
        v8f S[4];
#pragma unroll
        for (int j = 0; j < 4; ++j) S[j] = vzero;
        {
            H16 bfk[4][2];
#pragma unroll
            for (int j = 0; j < 4; ++j) {
                const int nB = j * 16 + colL;      // key column (s_local)
#pragma unroll
                for (int c = 0; c < 2; ++c) {
                    const int kbB = c * 32 + (hi ? 8 : 0);
                    bfk[j][c].u4[0] = *(const uint4*)&Ks[nB * 64 + kbB];
                    bfk[j][c].u4[1] = *(const uint4*)&Ks[nB * 64 + kbB + 16];
                }
            }
#pragma unroll
            for (int j = 0; j < 4; ++j) {
                S[j] = wmma_f16(qf[0].v, bfk[j][0].v, S[j]);
                S[j] = wmma_f16(qf[1].v, bfk[j][1].v, S[j]);
            }
        }

        // ---- masks + online softmax (rows live across 16 lanes of a half) ----
        bool padok[4];
#pragma unroll
        for (int j = 0; j < 4; ++j) {
            const int s = s0 + j * 16 + colL;
            padok[j] = (padmask[b * T_DIM + s] == 0);
        }
#pragma unroll
        for (int r = 0; r < 8; ++r) {
            const int qrow = q0 + r + (hi ? 8 : 0);
            float mx = -1e30f;
#pragma unroll
            for (int j = 0; j < 4; ++j) {
                const int s = s0 + j * 16 + colL;
                const float v = (padok[j] && s <= qrow) ? S[j][r] : -1e30f;
                S[j][r] = v;
                mx = fmaxf(mx, v);
            }
#pragma unroll
            for (int off = 8; off > 0; off >>= 1)
                mx = fmaxf(mx, __shfl_xor(mx, off, 16));
            const float mnew = fmaxf(m[r], mx);
            const float corr = __expf(m[r] - mnew);
            float rs = 0.f;
#pragma unroll
            for (int j = 0; j < 4; ++j) {
                const float p = __expf(S[j][r] - mnew);
                S[j][r] = p;
                rs += p;
            }
#pragma unroll
            for (int off = 8; off > 0; off >>= 1)
                rs += __shfl_xor(rs, off, 16);
            l[r] = l[r] * corr + rs;
            m[r] = mnew;
#pragma unroll
            for (int j = 0; j < 4; ++j) O[j][r] *= corr;
        }

        // ---- P: C-layout -> A-layout via per-wave LDS region ----
        half_t* myP = &Ps[wave * 16 * 64];
#pragma unroll
        for (int r = 0; r < 8; ++r) {
            const int rr = r + (hi ? 8 : 0);
#pragma unroll
            for (int j = 0; j < 4; ++j)
                myP[rr * 64 + j * 16 + colL] = (half_t)S[j][r];
        }
        H16 pf[2];
#pragma unroll
        for (int c = 0; c < 2; ++c) {
            pf[c].u4[0] = *(const uint4*)&myP[colL * 64 + c * 32 + kbA];
            pf[c].u4[1] = *(const uint4*)&myP[colL * 64 + c * 32 + kbA + 16];
        }

        // ---- O += P V: 8 B-frags, 8 WMMAs ----
        {
            H16 bfv[4][2];
#pragma unroll
            for (int j = 0; j < 4; ++j) {
                const int nV = j * 16 + colL;      // dd
#pragma unroll
                for (int c = 0; c < 2; ++c) {
                    const int kbB = c * 32 + (hi ? 8 : 0);
                    bfv[j][c].u4[0] = *(const uint4*)&Vs[nV * 64 + kbB];
                    bfv[j][c].u4[1] = *(const uint4*)&Vs[nV * 64 + kbB + 16];
                }
            }
#pragma unroll
            for (int j = 0; j < 4; ++j) {
                O[j] = wmma_f16(pf[0].v, bfv[j][0].v, O[j]);
                O[j] = wmma_f16(pf[1].v, bfv[j][1].v, O[j]);
            }
        }
        __syncthreads();
    }

    // ---- epilogue: normalize and scatter to [T,B,E] fp32 ----
    float linv[8];
#pragma unroll
    for (int r = 0; r < 8; ++r) linv[r] = 1.f / l[r];
#pragma unroll
    for (int j = 0; j < 4; ++j) {
        const int dd = j * 16 + colL;
        const int e  = h * D_DIM + dd;
#pragma unroll
        for (int r = 0; r < 8; ++r) {
            const int t = q0 + r + (hi ? 8 : 0);
            out[(size_t)(t * B_DIM + b) * E_DIM + e] = O[j][r] * linv[r];
        }
    }
}

// ---------------------------------------------------------------------------
extern "C" void kernel_launch(void* const* d_in, const int* in_sizes, int n_in,
                              void* d_out, int out_size, void* d_ws, size_t ws_size,
                              hipStream_t stream)
{
    (void)in_sizes; (void)n_in; (void)out_size; (void)ws_size;

    const float* query = (const float*)d_in[0];
    // d_in[1] = key (unused by reference self-attention path)
    const unsigned char* pad = (const unsigned char*)d_in[2];
    // d_in[3] = attn_mask (causal; applied analytically)
    const float* Wq = (const float*)d_in[4];
    const float* bq = (const float*)d_in[5];
    const float* Wk = (const float*)d_in[6];
    const float* bk = (const float*)d_in[7];
    const float* Wv = (const float*)d_in[8];
    const float* bv = (const float*)d_in[9];
    // d_in[10] = num_heads (hardcoded 16)

    // Workspace: Qh, Kh (row-major [bh][t][64]) + Vt (transposed [bh][dd][t]),
    // each 64*1024*64 fp16 = 8 MB -> 24 MB total.
    half_t* Qh = (half_t*)d_ws;
    half_t* Kh = Qh + (size_t)BH_DIM * T_DIM * D_DIM;
    half_t* Vt = Kh + (size_t)BH_DIM * T_DIM * D_DIM;

    const float scaling = 0.125f;   // D^-0.5, D=64

    dim3 g1(E_DIM / 64, ROWS / 128); // (16, 32)
    dim3 b1(256);
    qkv_proj_kernel<0><<<g1, b1, 0, stream>>>(query, Wq, bq, Qh, scaling);
    qkv_proj_kernel<0><<<g1, b1, 0, stream>>>(query, Wk, bk, Kh, 1.0f);
    qkv_proj_kernel<1><<<g1, b1, 0, stream>>>(query, Wv, bv, Vt, 1.0f);

    dim3 g2(BH_DIM, T_DIM / 64);    // (64, 16)
    attn_kernel<<<g2, 128, 0, stream>>>(Qh, Kh, Vt, pad, (float*)d_out);
}
